// GSR_52166672777965
// MI455X (gfx1250) — matrix operand order repaired and checked by reading, not compiled, not run on hardware
//
#include <hip/hip_runtime.h>
#include <hip/hip_bf16.h>
#include <math.h>
#include <stdint.h>

#define NEG_SLOPE 0.2f

typedef __attribute__((ext_vector_type(16))) __bf16 v16bf;
typedef __attribute__((ext_vector_type(8)))  __bf16 v8bf;
typedef __attribute__((ext_vector_type(8)))  float  v8f;
typedef __attribute__((ext_vector_type(4)))  int    v4i;

// -------- async global->LDS staging (gfx1250 path when toolchain exposes it)
__device__ __forceinline__ void lds_copy16(const __bf16* g, __bf16* l) {
#if __has_builtin(__builtin_amdgcn_global_load_async_to_lds_b128)
  typedef __attribute__((address_space(1))) v4i GV4;   // prints as "__device__"
  typedef __attribute__((address_space(3))) v4i LV4;   // prints as "__shared__"
  // flat LDS address truncates to the LDS byte offset (ISA 10.2 aperture rules)
  __builtin_amdgcn_global_load_async_to_lds_b128(
      (GV4*)(uintptr_t)g, (LV4*)(uint32_t)(uintptr_t)l, 0, 0);
#else
  *(v8bf*)l = *(const v8bf*)g;
#endif
}

__device__ __forceinline__ void lds_copy_fence() {
#if __has_builtin(__builtin_amdgcn_global_load_async_to_lds_b128)
#if __has_builtin(__builtin_amdgcn_s_wait_asynccnt)
  __builtin_amdgcn_s_wait_asynccnt(0);
#else
  asm volatile("s_wait_asynccnt 0" ::: "memory");
#endif
#endif
  __syncthreads();
}

// ---------------------------------------------------------------- utilities
__global__ void k_zero(float* __restrict__ p, int n) {
  int i = blockIdx.x * blockDim.x + threadIdx.x;
  if (i < n) p[i] = 0.0f;
}

// fp32 -> bf16 (row-major mirror of activations)
__global__ void k_cvt_bf16(const float* __restrict__ s, __bf16* __restrict__ d, int n) {
  int i = blockIdx.x * blockDim.x + threadIdx.x;
  if (i < n) d[i] = (__bf16)s[i];
}

// pack W[K x 256] fp32 -> Wp: for each ktile (32 K), col n: 32 contiguous bf16
// Wp[kt*8192 + n*32 + kk] = W[(kt*32+kk)*256 + n]
__global__ void k_pack_w(const float* __restrict__ W, __bf16* __restrict__ Wp, int total) {
  int i = blockIdx.x * blockDim.x + threadIdx.x;
  if (i >= total) return;
  int kt = i >> 13;
  int rem = i & 8191;
  int n = rem >> 5;
  int kk = rem & 31;
  Wp[i] = (__bf16)W[(size_t)(kt * 32 + kk) * 256 + n];
}

// ---------------------------------------------------------------- WMMA GEMM
// C[nrows x 256] = A[nrows x K] * W[K x 256]; A bf16 row-major, W packed (Wp).
// Block: 8 waves; wave -> 16x64 tile (4 x 16x16 acc). B slice staged in LDS.
// CDNA5 wave32 fragment layouts (ISA 7.12.2):
//   A 16x32: lanes0-15 M=lane, elems0-7=K0-7, 8-15=K16-23; lanes16-31 K+8
//   B 32x16: lanes0-15 N=lane, same K order as A; this code uses elem j=K 16*half+j for A and B
//   C 16x16: VGPR r -> M=r (lanes0-15) / M=8+r (lanes16-31), N=lane&15
__global__ __launch_bounds__(256) void k_gemm_wmma(
    const __bf16* __restrict__ A, const __bf16* __restrict__ Wp,
    float* __restrict__ C, int nrows, int K) {
  __shared__ __align__(64) __bf16 lds_w[8 * 64 * 32];   // up to K=256: 32 KB
  const int ktiles = K >> 5;
  const int ncol0  = blockIdx.y * 64;

  // cooperative stage of this block's 64-column slice of Wp into LDS
  // 16-byte units: per ktile 64 cols * 4 units
  const int n16 = ktiles * 256;
  for (int c = threadIdx.x; c < n16; c += 256) {
    int kt = c >> 8, rem = c & 255;
    int col = rem >> 2, q = rem & 3;
    const __bf16* g = Wp + ((size_t)(kt * 256 + ncol0 + col)) * 32 + q * 8;
    __bf16* l = lds_w + ((size_t)(kt * 64 + col)) * 32 + q * 8;
    lds_copy16(g, l);
  }
  lds_copy_fence();

  const int lane  = threadIdx.x & 31;
  const int wave  = threadIdx.x >> 5;
  const int mtile = blockIdx.x * 8 + wave;
  if (mtile * 16 >= nrows) return;              // uniform per wave, after barrier
  const int row0 = mtile * 16;
  const int half = lane >> 4;
  const int mr   = lane & 15;
  const __bf16* arow = A + (size_t)(row0 + mr) * K;

  v8f acc[4] = {};
  for (int kt = 0; kt < ktiles; ++kt) {
    const int k0 = kt * 32;
    v8bf alo = *(const v8bf*)(arow + k0 + half * 16);
    v8bf ahi = *(const v8bf*)(arow + k0 + half * 16 + 8);
    v16bf af;
#pragma unroll
    for (int j = 0; j < 8; ++j) { af[j] = alo[j]; af[8 + j] = ahi[j]; }
#pragma unroll
    for (int t = 0; t < 4; ++t) {
      const __bf16* lp = lds_w + ((size_t)(kt * 64 + t * 16 + mr)) * 32 + half * 16;
      v16bf bfr = *(const v16bf*)lp;            // 32B-aligned -> ds_load_b128 x2
      acc[t] = __builtin_amdgcn_wmma_f32_16x16x32_bf16(
          false, af, false, bfr, (short)0, acc[t], false, false);
    }
  }
#pragma unroll
  for (int t = 0; t < 4; ++t)
#pragma unroll
    for (int r = 0; r < 8; ++r)
      C[(size_t)(row0 + half * 8 + r) * 256 + ncol0 + t * 16 + mr] = acc[t][r];
}

// ---------------------------------------------- per-node attention half-logits
__global__ void k_attn_logits(const float* __restrict__ h,
                              const float* __restrict__ a_s,
                              const float* __restrict__ a_d,
                              float* __restrict__ al_s,
                              float* __restrict__ al_d, int n4) {
  int i = blockIdx.x * blockDim.x + threadIdx.x;
  if (i >= n4) return;
  int n = i >> 2, hd = i & 3;
  const float* hp  = h + (size_t)n * 256 + hd * 64;
  const float* asp = a_s + hd * 64;
  const float* adp = a_d + hd * 64;
  float ss = 0.f, sd = 0.f;
#pragma unroll 8
  for (int c = 0; c < 64; ++c) {
    float v = hp[c];
    ss += v * asp[c];
    sd += v * adp[c];
  }
  al_s[i] = ss;
  al_d[i] = sd;
}

// ---------------------------------------------- softmax denominator per (dst,h)
__global__ void k_edge_softmax_sum(const int* __restrict__ src,
                                   const int* __restrict__ dst,
                                   const float* __restrict__ al_s,
                                   const float* __restrict__ al_d,
                                   float* __restrict__ seg, int E, int tot4) {
  int i = blockIdx.x * blockDim.x + threadIdx.x;
  if (i >= tot4) return;
  int e = i >> 2, hd = i & 3;
  int s, d;
  if (e < E) { s = src[e]; d = dst[e]; } else { s = d = e - E; }  // self loops
  float l = al_s[s * 4 + hd] + al_d[d * 4 + hd];
  l = l > 0.f ? l : NEG_SLOPE * l;
  atomicAdd(&seg[d * 4 + hd], expf(l));
}

// ---------------------------------------------- weighted scatter-add per edge
__global__ __launch_bounds__(256) void k_edge_scatter(
    const int* __restrict__ src, const int* __restrict__ dst,
    const float* __restrict__ h, const float* __restrict__ al_s,
    const float* __restrict__ al_d, const float* __restrict__ seg,
    float* __restrict__ out, int E, int Etot) {
  int lane = threadIdx.x & 31;
  int e = blockIdx.x * 8 + (threadIdx.x >> 5);
  if (e >= Etot) return;                        // uniform per wave
  int s, d;
  if (e < E) { s = src[e]; d = dst[e]; } else { s = d = e - E; }
  int hd = lane >> 3;
  float l = al_s[s * 4 + hd] + al_d[d * 4 + hd];
  l = l > 0.f ? l : NEG_SLOPE * l;
  float alpha = expf(l) / seg[d * 4 + hd];
  int c0 = lane * 8;
  const float4* hs = (const float4*)(h + (size_t)s * 256 + c0);
  float4 v0 = hs[0], v1 = hs[1];
  float* op = out + (size_t)d * 256 + c0;
  atomicAdd(op + 0, v0.x * alpha); atomicAdd(op + 1, v0.y * alpha);
  atomicAdd(op + 2, v0.z * alpha); atomicAdd(op + 3, v0.w * alpha);
  atomicAdd(op + 4, v1.x * alpha); atomicAdd(op + 5, v1.y * alpha);
  atomicAdd(op + 6, v1.z * alpha); atomicAdd(op + 7, v1.w * alpha);
}

// ---------------------------------------------- bias + tanh epilogue (+bf16 mirror)
__global__ void k_bias_tanh_cvt(float* __restrict__ out, __bf16* __restrict__ xb,
                                const float* __restrict__ b, int total) {
  int i = blockIdx.x * blockDim.x + threadIdx.x;
  if (i >= total) return;
  float v = tanhf(out[i] + b[i & 255]);
  out[i] = v;
  xb[i] = (__bf16)v;
}

// ---------------------------------------------- layer 5 (256 -> 1, H=1)
__global__ void k_h5(const float* __restrict__ x, const float* __restrict__ W,
                     float* __restrict__ h5, int N) {
  int n = blockIdx.x * blockDim.x + threadIdx.x;
  if (n >= N) return;
  const float* xr = x + (size_t)n * 256;
  float acc = 0.f;
#pragma unroll 8
  for (int k = 0; k < 256; ++k) acc += xr[k] * W[k];
  h5[n] = acc;
}

__global__ void k_edge5_sum(const int* __restrict__ src, const int* __restrict__ dst,
                            const float* __restrict__ h5, const float* __restrict__ as,
                            const float* __restrict__ ad, float* __restrict__ seg,
                            int E, int Etot) {
  int e = blockIdx.x * blockDim.x + threadIdx.x;
  if (e >= Etot) return;
  int s, d;
  if (e < E) { s = src[e]; d = dst[e]; } else { s = d = e - E; }
  float l = h5[s] * as[0] + h5[d] * ad[0];
  l = l > 0.f ? l : NEG_SLOPE * l;
  atomicAdd(&seg[d], expf(l));
}

__global__ void k_edge5_scatter(const int* __restrict__ src, const int* __restrict__ dst,
                                const float* __restrict__ h5, const float* __restrict__ as,
                                const float* __restrict__ ad, const float* __restrict__ seg,
                                float* __restrict__ nodeout, int E, int Etot) {
  int e = blockIdx.x * blockDim.x + threadIdx.x;
  if (e >= Etot) return;
  int s, d;
  if (e < E) { s = src[e]; d = dst[e]; } else { s = d = e - E; }
  float l = h5[s] * as[0] + h5[d] * ad[0];
  l = l > 0.f ? l : NEG_SLOPE * l;
  float alpha = expf(l) / seg[d];
  atomicAdd(&nodeout[d], h5[s] * alpha);
}

__global__ void k_pool(float* __restrict__ nodeout, const float* __restrict__ b5,
                       const int* __restrict__ batch, float* __restrict__ gsum,
                       float* __restrict__ gcnt, int N) {
  int n = blockIdx.x * blockDim.x + threadIdx.x;
  if (n >= N) return;
  float v = nodeout[n] + b5[0];
  nodeout[n] = v;                 // final node_preds
  int g = batch[n];
  atomicAdd(&gsum[g], v);
  atomicAdd(&gcnt[g], 1.0f);
}

__global__ void k_graph_out(const float* __restrict__ gsum,
                            const float* __restrict__ gcnt,
                            float* __restrict__ gout, int G) {
  int g = blockIdx.x * blockDim.x + threadIdx.x;
  if (g >= G) return;
  gout[g] = gsum[g] / fmaxf(gcnt[g], 1.0f);
}

// ---------------------------------------------------------------- launcher
extern "C" void kernel_launch(void* const* d_in, const int* in_sizes, int n_in,
                              void* d_out, int out_size, void* d_ws, size_t ws_size,
                              hipStream_t stream) {
  const float* x     = (const float*)d_in[0];
  const int*   ei    = (const int*)d_in[1];
  const int*   batch = (const int*)d_in[2];

  const int N    = in_sizes[0] / 128;   // 50000
  const int E    = in_sizes[1] / 2;     // 800000
  const int Etot = E + N;               // + self loops
  const int G    = 512;
  const int* src = ei;
  const int* dst = ei + E;

  // workspace carve-up
  float* bufH = (float*)d_ws;                      // [N,256] GEMM output h
  float* bufX = bufH + (size_t)N * 256;            // [N,256] aggregation / next x
  float* al_s = bufX + (size_t)N * 256;            // [N,4]
  float* al_d = al_s + (size_t)N * 4;              // [N,4]
  float* seg  = al_d + (size_t)N * 4;              // [N,4]
  float* h5   = seg  + (size_t)N * 4;              // [N]
  float* seg5 = h5   + N;                          // [N]
  float* gsum = seg5 + N;                          // [G]
  float* gcnt = gsum + G;                          // [G]
  __bf16* xb  = (__bf16*)(gcnt + G);               // [N,256] bf16 activations
  __bf16* Wp  = xb + (size_t)N * 256;              // [256*256] packed weights

  float* nodeout = (float*)d_out;                  // [N]  node_preds
  float* gout    = nodeout + N;                    // [G]  graph_preds

  const int TB = 256;
  auto cdiv = [](int a, int b) { return (a + b - 1) / b; };

  // bf16 mirror of the input features (layer-1 A operand, K=128)
  k_cvt_bf16<<<cdiv(N * 128, TB), TB, 0, stream>>>(x, xb, N * 128);

  int K = 128;
  for (int l = 0; l < 4; ++l) {
    const float* W  = (const float*)d_in[3 + 4 * l + 0];
    const float* as = (const float*)d_in[3 + 4 * l + 1];
    const float* ad = (const float*)d_in[3 + 4 * l + 2];
    const float* b  = (const float*)d_in[3 + 4 * l + 3];

    k_pack_w<<<cdiv(K * 256, TB), TB, 0, stream>>>(W, Wp, K * 256);
    dim3 ggrid(cdiv(N / 16, 8), 4);
    k_gemm_wmma<<<ggrid, TB, 0, stream>>>(xb, Wp, bufH, N, K);
    k_attn_logits<<<cdiv(N * 4, TB), TB, 0, stream>>>(bufH, as, ad, al_s, al_d, N * 4);
    k_zero<<<cdiv(N * 4, TB), TB, 0, stream>>>(seg, N * 4);
    k_zero<<<cdiv(N * 256, TB), TB, 0, stream>>>(bufX, N * 256);
    k_edge_softmax_sum<<<cdiv(Etot * 4, TB), TB, 0, stream>>>(src, dst, al_s, al_d,
                                                              seg, E, Etot * 4);
    k_edge_scatter<<<cdiv(Etot, 8), TB, 0, stream>>>(src, dst, bufH, al_s, al_d,
                                                     seg, bufX, E, Etot);
    k_bias_tanh_cvt<<<cdiv(N * 256, TB), TB, 0, stream>>>(bufX, xb, b, N * 256);
    K = 256;
  }

  // final layer (256 -> 1, single head, no tanh)
  const float* W5  = (const float*)d_in[19];
  const float* as5 = (const float*)d_in[20];
  const float* ad5 = (const float*)d_in[21];
  const float* b5  = (const float*)d_in[22];

  k_h5<<<cdiv(N, TB), TB, 0, stream>>>(bufX, W5, h5, N);
  k_zero<<<cdiv(N, TB), TB, 0, stream>>>(seg5, N);
  k_zero<<<cdiv(N, TB), TB, 0, stream>>>(nodeout, N);
  k_zero<<<cdiv(G, TB), TB, 0, stream>>>(gsum, G);
  k_zero<<<cdiv(G, TB), TB, 0, stream>>>(gcnt, G);
  k_edge5_sum<<<cdiv(Etot, TB), TB, 0, stream>>>(src, dst, h5, as5, ad5, seg5, E, Etot);
  k_edge5_scatter<<<cdiv(Etot, TB), TB, 0, stream>>>(src, dst, h5, as5, ad5, seg5,
                                                     nodeout, E, Etot);
  k_pool<<<cdiv(N, TB), TB, 0, stream>>>(nodeout, b5, batch, gsum, gcnt, N);
  k_graph_out<<<cdiv(G, TB), TB, 0, stream>>>(gsum, gcnt, gout, G);
}
